// GPT_85366769975919
// MI455X (gfx1250) — compile-verified, matrix operand order checked by reading, not run on hardware
//
#include <hip/hip_runtime.h>

// CDNA5 / gfx1250, wave32. All GEMMs run on v_wmma_f32_16x16x32_bf16 with
// fp32->bf16 conversion at the LDS staging step. The model is compute-bound
// (~615 GFLOP vs ~550 MB -> ~24us HBM time at 23.3 TB/s), so staging is
// double-buffered in LDS (1 barrier per K-step) to keep the WMMA pipe busy.

typedef __attribute__((ext_vector_type(16))) __bf16 v16bf;
typedef __attribute__((ext_vector_type(8)))  float  v8f;

constexpr int Bc = 2, Tc = 1024, Ec = 1024, Hc = 16, Dc = 64, Lc = 8, Vc = 32000;
constexpr int BT = Bc * Tc;

// 16-bit A/B fragment K-offset pattern (ISA 7.12.2, 16x32 bf16):
// lanes 0-15 hold K {0..7,16..23}, lanes 16-31 hold K {8..15,24..31}.
__device__ __forceinline__ int koff16(int jj, int kbase) {
  return (jj & 7) + ((jj >> 3) << 4) + kbase;
}

// ----------------------------- embedding -----------------------------------
__global__ __launch_bounds__(256) void embed_kernel(
    const int* __restrict__ idx, const float* __restrict__ tok,
    const float* __restrict__ pos, float* __restrict__ x)
{
  const int row = blockIdx.x;            // b*T + t
  const int t = row & (Tc - 1);
  const int token = idx[row];
  const float* te = tok + (size_t)token * Ec;
  const float* pe = pos + (size_t)t * Ec;
  float* xr = x + (size_t)row * Ec;
  #pragma unroll
  for (int i = 0; i < 4; ++i) {
    int c = threadIdx.x + i * 256;
    xr[c] = te[c] + pe[c];
  }
}

// ----------------------------- layernorm -----------------------------------
__global__ __launch_bounds__(256) void layernorm_kernel(
    const float* __restrict__ x, const float* __restrict__ gamma,
    const float* __restrict__ beta, float* __restrict__ out)
{
  const int row = blockIdx.x;
  const float* xr = x + (size_t)row * Ec;
  float v[4]; float s = 0.f, ss = 0.f;
  #pragma unroll
  for (int i = 0; i < 4; ++i) {
    v[i] = xr[threadIdx.x + i * 256];
    s += v[i]; ss += v[i] * v[i];
  }
  #pragma unroll
  for (int off = 16; off >= 1; off >>= 1) {
    s  += __shfl_xor(s,  off, 32);
    ss += __shfl_xor(ss, off, 32);
  }
  __shared__ float shs[8], shq[8], stat[2];
  const int wave = threadIdx.x >> 5, lane = threadIdx.x & 31;
  if (lane == 0) { shs[wave] = s; shq[wave] = ss; }
  __syncthreads();
  if (threadIdx.x == 0) {
    float a = 0.f, b = 0.f;
    #pragma unroll
    for (int w = 0; w < 8; ++w) { a += shs[w]; b += shq[w]; }
    float mu = a * (1.f / Ec);
    float var = b * (1.f / Ec) - mu * mu;
    stat[0] = mu; stat[1] = rsqrtf(var + 1e-5f);
  }
  __syncthreads();
  const float mu = stat[0], rstd = stat[1];
  float* outr = out + (size_t)row * Ec;
  #pragma unroll
  for (int i = 0; i < 4; ++i) {
    int c = threadIdx.x + i * 256;
    outr[c] = (v[i] - mu) * rstd * gamma[c] + beta[c];
  }
}

// ------------------------------- GEMM ---------------------------------------
// C[M,N] = A[M,K] @ B[N,K]^T, EPI: 0 = none, 1 = += residual R, 2 = exact GELU.
// 128x128 block tile, 8 waves in a 2x4 grid, each wave owns a 64x32 tile
// (4x2 WMMA fragments). Double-buffered LDS, one barrier per 32-wide K-step.
template <int EPI>
__global__ __launch_bounds__(256) void gemm_kernel(
    const float* __restrict__ A, const float* __restrict__ Bw,
    const float* __restrict__ R, float* __restrict__ C,
    int M, int N, int K)
{
  constexpr int LDK = 34;                 // padded halves -> conflict-free
  __shared__ __bf16 As[2][128 * LDK];
  __shared__ __bf16 Bs[2][128 * LDK];

  const int tid   = threadIdx.x;
  const int lane  = tid & 31;
  const int wave  = tid >> 5;
  const int waveM = wave >> 2;            // 0..1
  const int waveN = wave & 3;             // 0..3
  const int halfSel = lane >> 4;
  const int kbase   = halfSel << 3;
  const int lrow    = lane & 15;
  const int m0 = blockIdx.y * 128;
  const int n0 = blockIdx.x * 128;

  v8f acc[4][2];
  #pragma unroll
  for (int i = 0; i < 4; ++i)
    #pragma unroll
    for (int j = 0; j < 2; ++j)
      #pragma unroll
      for (int r = 0; r < 8; ++r) acc[i][j][r] = 0.f;

  float4 aReg[4], bReg[4];
  auto loadRegs = [&](int kt) {
    #pragma unroll
    for (int it = 0; it < 4; ++it) {
      int f = tid + it * 256;             // 1024 float4 per tile
      int rw = f >> 3;
      int kc = (f & 7) << 2;
      aReg[it] = *(const float4*)(A  + (size_t)(m0 + rw) * K + kt + kc);
      bReg[it] = *(const float4*)(Bw + (size_t)(n0 + rw) * K + kt + kc);
    }
  };
  auto storeLDS = [&](int buf) {
    #pragma unroll
    for (int it = 0; it < 4; ++it) {
      int f = tid + it * 256;
      int rw = f >> 3;
      int kc = (f & 7) << 2;
      __bf16* ap = As[buf] + rw * LDK + kc;
      ap[0] = (__bf16)aReg[it].x; ap[1] = (__bf16)aReg[it].y;
      ap[2] = (__bf16)aReg[it].z; ap[3] = (__bf16)aReg[it].w;
      __bf16* bp = Bs[buf] + rw * LDK + kc;
      bp[0] = (__bf16)bReg[it].x; bp[1] = (__bf16)bReg[it].y;
      bp[2] = (__bf16)bReg[it].z; bp[3] = (__bf16)bReg[it].w;
    }
  };

  loadRegs(0);
  storeLDS(0);
  __syncthreads();
  int cur = 0;

  for (int kt = 0; kt < K; kt += 32) {
    const bool hasNext = (kt + 32) < K;
    if (hasNext) loadRegs(kt + 32);       // global loads overlap compute

    const __bf16* Ab = As[cur];
    const __bf16* Bb = Bs[cur];
    v16bf af[4], bfm[2];
    #pragma unroll
    for (int mt = 0; mt < 4; ++mt) {
      const __bf16* p = Ab + (waveM * 64 + mt * 16 + lrow) * LDK;
      #pragma unroll
      for (int jj = 0; jj < 16; ++jj) af[mt][jj] = p[koff16(jj, kbase)];
    }
    #pragma unroll
    for (int nt = 0; nt < 2; ++nt) {
      const __bf16* p = Bb + (waveN * 32 + nt * 16 + lrow) * LDK;
      #pragma unroll
      for (int jj = 0; jj < 16; ++jj) bfm[nt][jj] = p[koff16(jj, kbase)];
    }
    #pragma unroll
    for (int mt = 0; mt < 4; ++mt)
      #pragma unroll
      for (int nt = 0; nt < 2; ++nt)
        acc[mt][nt] = __builtin_amdgcn_wmma_f32_16x16x32_bf16(
            false, af[mt], false, bfm[nt], (short)0, acc[mt][nt], false, false);

    if (hasNext) {
      storeLDS(cur ^ 1);                  // safe: cur^1 last read before prev barrier
      __syncthreads();
      cur ^= 1;
    }
  }

  // ---- epilogue: C layout => VGPR r holds row r + 8*halfSel, col = lrow ----
  #pragma unroll
  for (int mt = 0; mt < 4; ++mt) {
    #pragma unroll
    for (int nt = 0; nt < 2; ++nt) {
      const int col = n0 + waveN * 32 + nt * 16 + lrow;
      #pragma unroll
      for (int r = 0; r < 8; ++r) {
        const int rowC = m0 + waveM * 64 + mt * 16 + kbase + r;
        float vv = acc[mt][nt][r];
        if constexpr (EPI == 1) vv += R[(size_t)rowC * N + col];
        if constexpr (EPI == 2) vv = 0.5f * vv * (1.f + erff(vv * 0.70710678118654752f));
        C[(size_t)rowC * N + col] = vv;
      }
    }
  }
}

// --------------------------- flash attention --------------------------------
// grid = (T/128, B*H). One block: 128 q-rows, 8 waves x 16 rows.
// Streaming softmax over 32-key chunks; S and P·V on bf16 WMMA.
// K/V staging double-buffered in LDS, one barrier per chunk.
__global__ __launch_bounds__(256) void attention_kernel(
    const float* __restrict__ qkv, float* __restrict__ y)
{
  constexpr int E3 = 3 * Ec;
  constexpr int KS = 66;                 // Ks/Vs row stride (halves), odd banks
  constexpr int PS = 34;                 // P staging stride
  __shared__ __bf16 Ks[2][32 * KS];
  __shared__ __bf16 Vs[2][32 * KS];
  __shared__ __bf16 Ps[8][16 * PS];

  const int tid   = threadIdx.x;
  const int lane  = tid & 31;
  const int wave  = tid >> 5;
  const int halfSel = lane >> 4;
  const int kbase   = halfSel << 3;
  const int lrow    = lane & 15;

  const int bh = blockIdx.y;
  const int b  = bh >> 4;                // H = 16
  const int h  = bh & 15;
  const int q0 = blockIdx.x * 128 + wave * 16;

  // Q fragment: 16 rows x 64 dims -> two 16x32 bf16 A-fragments
  v16bf qf[2];
  {
    const float* qp = qkv + (size_t)(b * Tc + q0 + lrow) * E3 + h * Dc;
    #pragma unroll
    for (int kc = 0; kc < 2; ++kc)
      #pragma unroll
      for (int jj = 0; jj < 16; ++jj)
        qf[kc][jj] = (__bf16)qp[kc * 32 + koff16(jj, kbase)];
  }

  v8f o[4];
  #pragma unroll
  for (int d = 0; d < 4; ++d)
    #pragma unroll
    for (int r = 0; r < 8; ++r) o[d][r] = 0.f;
  float rm[8], rl[8];
  #pragma unroll
  for (int r = 0; r < 8; ++r) { rm[r] = -1e30f; rl[r] = 0.f; }

  const int jEnd = blockIdx.x * 128 + 128;
  const float scale = 0.125f;            // 1/sqrt(64)
  __bf16* Pw = Ps[wave];

  float4 kReg[2], vReg[2];
  auto loadKV = [&](int j) {
    #pragma unroll
    for (int it = 0; it < 2; ++it) {
      int f  = tid + it * 256;           // 512 float4 per matrix
      int kr = f >> 4;
      int dc = (f & 15) << 2;
      const float* kp = qkv + (size_t)(b * Tc + j + kr) * E3 + Ec + h * Dc + dc;
      kReg[it] = *(const float4*)kp;
      vReg[it] = *(const float4*)(kp + Ec);
    }
  };
  auto storeKV = [&](int buf) {
    #pragma unroll
    for (int it = 0; it < 2; ++it) {
      int f  = tid + it * 256;
      int kr = f >> 4;
      int dc = (f & 15) << 2;
      __bf16* kd = Ks[buf] + kr * KS + dc;
      kd[0] = (__bf16)kReg[it].x; kd[1] = (__bf16)kReg[it].y;
      kd[2] = (__bf16)kReg[it].z; kd[3] = (__bf16)kReg[it].w;
      __bf16* vd = Vs[buf] + kr * KS + dc;
      vd[0] = (__bf16)vReg[it].x; vd[1] = (__bf16)vReg[it].y;
      vd[2] = (__bf16)vReg[it].z; vd[3] = (__bf16)vReg[it].w;
    }
  };

  loadKV(0);
  storeKV(0);
  __syncthreads();
  int cur = 0;

  for (int j = 0; j < jEnd; j += 32) {
    const bool hasNext = (j + 32) < jEnd;
    if (hasNext) loadKV(j + 32);         // overlap next chunk's global loads

    if (j <= q0 + 15) {                  // wave-uniform: EXEC all-ones at WMMA
      const __bf16* Kb = Ks[cur];
      const __bf16* Vb = Vs[cur];
      // S = Q @ K^T : two 16x16 N-tiles, D=64 -> 2 K-steps each
      v8f s[2];
      #pragma unroll
      for (int nt = 0; nt < 2; ++nt)
        #pragma unroll
        for (int r = 0; r < 8; ++r) s[nt][r] = 0.f;
      #pragma unroll
      for (int nt = 0; nt < 2; ++nt) {
        #pragma unroll
        for (int kc = 0; kc < 2; ++kc) {
          v16bf kf;
          const __bf16* p = Kb + (nt * 16 + lrow) * KS + kc * 32;
          #pragma unroll
          for (int jj = 0; jj < 16; ++jj) kf[jj] = p[koff16(jj, kbase)];
          s[nt] = __builtin_amdgcn_wmma_f32_16x16x32_bf16(
              false, qf[kc], false, kf, (short)0, s[nt], false, false);
        }
      }
      // causal mask + online softmax (rows live across 16 lanes of a half)
      #pragma unroll
      for (int r = 0; r < 8; ++r) {
        const int rowT = q0 + kbase + r;
        float s0 = s[0][r] * scale;
        float s1 = s[1][r] * scale;
        if (j + lrow      > rowT) s0 = -1e30f;
        if (j + 16 + lrow > rowT) s1 = -1e30f;
        float mx = fmaxf(s0, s1);
        #pragma unroll
        for (int off = 8; off >= 1; off >>= 1)
          mx = fmaxf(mx, __shfl_xor(mx, off, 32));
        const float mnew = fmaxf(rm[r], mx);
        const float corr = __expf(rm[r] - mnew);
        const float e0 = __expf(s0 - mnew);
        const float e1 = __expf(s1 - mnew);
        float rs = e0 + e1;
        #pragma unroll
        for (int off = 8; off >= 1; off >>= 1)
          rs += __shfl_xor(rs, off, 32);
        rl[r] = rl[r] * corr + rs;
        rm[r] = mnew;
        #pragma unroll
        for (int d = 0; d < 4; ++d) o[d][r] *= corr;
        Pw[(kbase + r) * PS + lrow]      = (__bf16)e0;
        Pw[(kbase + r) * PS + 16 + lrow] = (__bf16)e1;
      }
      // same-wave LDS RAW: DS ops are in-order, wait before consuming
      asm volatile("s_wait_dscnt 0" ::: "memory");
      // P (16x32) as A-fragment, V (32x64) as 4 B-fragments
      v16bf pf;
      #pragma unroll
      for (int jj = 0; jj < 16; ++jj)
        pf[jj] = Pw[lrow * PS + koff16(jj, kbase)];
      #pragma unroll
      for (int d = 0; d < 4; ++d) {
        v16bf vf;
        #pragma unroll
        for (int jj = 0; jj < 16; ++jj)
          vf[jj] = Vb[koff16(jj, kbase) * KS + d * 16 + lrow];
        o[d] = __builtin_amdgcn_wmma_f32_16x16x32_bf16(
            false, pf, false, vf, (short)0, o[d], false, false);
      }
    }

    if (hasNext) {
      storeKV(cur ^ 1);
      __syncthreads();
      cur ^= 1;
    }
  }

  #pragma unroll
  for (int r = 0; r < 8; ++r) {
    const float inv = 1.f / rl[r];
    const int t = q0 + kbase + r;
    float* yr = y + (size_t)(b * Tc + t) * Ec + h * Dc;
    #pragma unroll
    for (int d = 0; d < 4; ++d)
      yr[d * 16 + lrow] = o[d][r] * inv;
  }
}

// ------------------------------- driver -------------------------------------
extern "C" void kernel_launch(void* const* d_in, const int* in_sizes, int n_in,
                              void* d_out, int out_size, void* d_ws, size_t ws_size,
                              hipStream_t stream) {
  (void)in_sizes; (void)n_in; (void)out_size; (void)ws_size;
  const int*   idx         = (const int*)  d_in[0];
  const float* tok_emb     = (const float*)d_in[1];
  const float* pos_emb     = (const float*)d_in[2];
  const float* qkv_w       = (const float*)d_in[3];
  const float* attn_proj_w = (const float*)d_in[4];
  const float* fc_w        = (const float*)d_in[5];
  const float* fc_proj_w   = (const float*)d_in[6];
  const float* ln1_s       = (const float*)d_in[7];
  const float* ln1_b       = (const float*)d_in[8];
  const float* ln2_s       = (const float*)d_in[9];
  const float* ln2_b       = (const float*)d_in[10];
  const float* lnf_s       = (const float*)d_in[11];
  const float* lnf_b       = (const float*)d_in[12];

  float* ws  = (float*)d_ws;
  float* x   = ws;                              // [BT, E]
  float* h   = x   + (size_t)BT * Ec;           // [BT, E]
  float* qkv = h   + (size_t)BT * Ec;           // [BT, 3E]
  float* y   = qkv + (size_t)BT * 3 * Ec;       // [BT, E]
  float* mfc = y   + (size_t)BT * Ec;           // [BT, 4E]

  embed_kernel<<<BT, 256, 0, stream>>>(idx, tok_emb, pos_emb, x);

  for (int l = 0; l < Lc; ++l) {
    layernorm_kernel<<<BT, 256, 0, stream>>>(x, ln1_s + (size_t)l * Ec,
                                             ln1_b + (size_t)l * Ec, h);
    gemm_kernel<0><<<dim3(3 * Ec / 128, BT / 128), 256, 0, stream>>>(
        h, qkv_w + (size_t)l * 3 * Ec * Ec, nullptr, qkv, BT, 3 * Ec, Ec);
    attention_kernel<<<dim3(Tc / 128, Bc * Hc), 256, 0, stream>>>(qkv, y);
    gemm_kernel<1><<<dim3(Ec / 128, BT / 128), 256, 0, stream>>>(
        y, attn_proj_w + (size_t)l * Ec * Ec, x, x, BT, Ec, Ec);
    layernorm_kernel<<<BT, 256, 0, stream>>>(x, ln2_s + (size_t)l * Ec,
                                             ln2_b + (size_t)l * Ec, h);
    gemm_kernel<2><<<dim3(4 * Ec / 128, BT / 128), 256, 0, stream>>>(
        h, fc_w + (size_t)l * 4 * Ec * Ec, nullptr, mfc, BT, 4 * Ec, Ec);
    gemm_kernel<1><<<dim3(Ec / 128, BT / 128), 256, 0, stream>>>(
        mfc, fc_proj_w + (size_t)l * Ec * 4 * Ec, x, x, BT, Ec, 4 * Ec);
  }

  layernorm_kernel<<<BT, 256, 0, stream>>>(x, lnf_s, lnf_b, h);
  gemm_kernel<0><<<dim3(Vc / 128, BT / 128), 256, 0, stream>>>(
      h, tok_emb, nullptr, (float*)d_out, BT, Vc, Ec);
}
